// HMADFramework_73366631350925
// MI455X (gfx1250) — matrix operand order repaired and checked by reading, not compiled, not run on hardware
//
#include <hip/hip_runtime.h>

typedef __attribute__((ext_vector_type(16))) _Float16 v16h;
typedef __attribute__((ext_vector_type(8)))  _Float16 v8h;
typedef __attribute__((ext_vector_type(8)))  float    v8f;

#define B_DIM 128
#define C_DIM 128
#define T_DIM 2048
#define E_DIM 64

// ---------------------------------------------------------------------------
// Kernel 1: Hilbert phase -> unit phasors z = analytic/|analytic| as f16 cos/sin.
// Stockham self-sorting radix-2 FFT in LDS (no bit-reversal), forward + filter
// + inverse. The IFFT 1/N scale cancels in the |a| normalization.
// ---------------------------------------------------------------------------
__device__ __forceinline__ void fft_pass(float sgn,
                                         const float* __restrict__ sr,
                                         const float* __restrict__ si,
                                         float* __restrict__ dr,
                                         float* __restrict__ di,
                                         int logs, int n) {
  const int s    = 1 << logs;
  const int half = n >> 1;
  for (int t = threadIdx.x; t < (T_DIM >> 1); t += 256) {
    int q = t & (s - 1);
    int p = t >> logs;
    int ia = q + s * p;
    int ib = ia + s * half;
    float ar = sr[ia], ai = si[ia];
    float br = sr[ib], bi = si[ib];
    float wr, wi;
    __sincosf(sgn * 6.28318530717958647692f * (float)p / (float)n, &wi, &wr);
    float tr = ar - br, ti = ai - bi;
    int o0 = q + ((2 * p) << logs);
    dr[o0]     = ar + br;
    di[o0]     = ai + bi;
    dr[o0 + s] = tr * wr - ti * wi;
    di[o0 + s] = tr * wi + ti * wr;
  }
}

__global__ void __launch_bounds__(256) hilbert_kernel(const float* __restrict__ x,
                                                      _Float16* __restrict__ zc,
                                                      _Float16* __restrict__ zs) {
  __shared__ float reA[T_DIM], imA[T_DIM], reB[T_DIM], imB[T_DIM];
  const size_t row = blockIdx.x;  // b*C + c
  const float* xin = x + row * T_DIM;
  for (int i = threadIdx.x; i < T_DIM; i += 256) { reA[i] = xin[i]; imA[i] = 0.0f; }

  // ping-pong pointers (runtime values; no static addrspacecast initializers)
  float* srcR = reA;
  float* srcI = imA;
  float* dstR = reB;
  float* dstI = imB;
  int n = T_DIM;
  for (int st = 0; st < 11; ++st) {         // forward FFT
    __syncthreads();
    fft_pass(-1.0f, srcR, srcI, dstR, dstI, st, n);
    float* t0 = srcR; srcR = dstR; dstR = t0;
    float* t1 = srcI; srcI = dstI; dstI = t1;
    n >>= 1;
  }
  __syncthreads();
  for (int i = threadIdx.x; i < T_DIM; i += 256) {   // analytic-signal filter
    float g = (i == 0 || i == (T_DIM / 2)) ? 1.0f : ((i < T_DIM / 2) ? 2.0f : 0.0f);
    srcR[i] *= g;
    srcI[i] *= g;
  }
  n = T_DIM;
  for (int st = 0; st < 11; ++st) {         // inverse FFT (conjugate twiddles)
    __syncthreads();
    fft_pass(1.0f, srcR, srcI, dstR, dstI, st, n);
    float* t0 = srcR; srcR = dstR; dstR = t0;
    float* t1 = srcI; srcI = dstI; dstI = t1;
    n >>= 1;
  }
  __syncthreads();
  _Float16* zcr = zc + row * T_DIM;
  _Float16* zsr = zs + row * T_DIM;
  for (int i = threadIdx.x; i < T_DIM; i += 256) {
    float re = srcR[i], im = srcI[i];
    float inv = rsqrtf(re * re + im * im + 1e-30f);
    zcr[i] = (_Float16)(re * inv);
    zsr[i] = (_Float16)(im * inv);
  }
}

// ---------------------------------------------------------------------------
// WMMA tile loaders (16-bit A 16x32, B 32x16), same contiguous k order for A and B.
// A: lane L -> row i0+(L&15), 16 contiguous k (lanes>=16 at k0+16).
// B: lane L    -> column j0+(L&15), 16 contiguous k (lanes>=16 at k0+16),
//    loaded from a transposed (row-major N x K) image so loads are contiguous.
// ---------------------------------------------------------------------------
__device__ __forceinline__ v16h load_a_tile(const _Float16* __restrict__ base,
                                            int row0, int k0, int ld) {
  int lane = threadIdx.x & 31;
  const _Float16* p = base + (size_t)(row0 + (lane & 15)) * ld
                           + k0 + ((lane & 16) ? 16 : 0);
  return *(const v16h*)(p);
}

__device__ __forceinline__ v16h load_b_tile(const _Float16* __restrict__ baseT,
                                            int n0, int k0, int ldT) {
  int lane = threadIdx.x & 31;
  const _Float16* p = baseT + (size_t)(n0 + (lane & 15)) * ldT
                            + k0 + ((lane & 16) ? 16 : 0);
  return *(const v16h*)(p);
}

// ---------------------------------------------------------------------------
// Kernel 2: connectivity. PSI tile via 4 WMMA accumulations per 32-wide k-step:
//   Re = C.C^T + S.S^T ; Im = S.C^T - C.S^T ; conn = sqrt(Re^2+Im^2)/T.
// One wave per 16x16 output tile; 8 waves per block.
// ---------------------------------------------------------------------------
__global__ void __launch_bounds__(256) conn_kernel(const _Float16* __restrict__ zc,
                                                   const _Float16* __restrict__ zs,
                                                   float* __restrict__ conn_out,
                                                   _Float16* __restrict__ conn_h) {
  int wid  = (blockIdx.x << 3) + (threadIdx.x >> 5);
  int b    = wid >> 6;          // 64 tiles per batch (8x8)
  int tile = wid & 63;
  int i0 = (tile >> 3) << 4;
  int j0 = (tile & 7) << 4;
  const _Float16* zcb = zc + (size_t)b * C_DIM * T_DIM;
  const _Float16* zsb = zs + (size_t)b * C_DIM * T_DIM;

  v8f accRE = {}, accSC = {}, accCS = {};
  for (int k0 = 0; k0 < T_DIM; k0 += 32) {
    v16h ca = load_a_tile(zcb, i0, k0, T_DIM);
    v16h sa = load_a_tile(zsb, i0, k0, T_DIM);
    v16h cb = load_b_tile(zcb, j0, k0, T_DIM);
    v16h sb = load_b_tile(zsb, j0, k0, T_DIM);
    accRE = __builtin_amdgcn_wmma_f32_16x16x32_f16(false, ca, false, cb, (short)0, accRE, false, false);
    accRE = __builtin_amdgcn_wmma_f32_16x16x32_f16(false, sa, false, sb, (short)0, accRE, false, false);
    accSC = __builtin_amdgcn_wmma_f32_16x16x32_f16(false, sa, false, cb, (short)0, accSC, false, false);
    accCS = __builtin_amdgcn_wmma_f32_16x16x32_f16(false, ca, false, sb, (short)0, accCS, false, false);
  }
  int lane = threadIdx.x & 31;
  int col  = j0 + (lane & 15);
  int rb   = i0 + ((lane & 16) ? 8 : 0);
#pragma unroll
  for (int v = 0; v < 8; ++v) {
    int row  = rb + v;
    float re = accRE[v];
    float im = accSC[v] - accCS[v];
    float c  = sqrtf(re * re + im * im) * (1.0f / (float)T_DIM);
    if (row == col) c = 0.0f;
    size_t idx = ((size_t)b * C_DIM + row) * C_DIM + col;
    conn_out[idx] = c;
    conn_h[idx]   = (_Float16)c;
  }
}

// ---------------------------------------------------------------------------
// Generic batched 16x16-tile f16 WMMA GEMM: D = A(row-major) x B (via B^T
// row-major) + bias, optional relu; writes f32 and/or f16 and/or f16-transposed.
// One wave per tile, 8 waves per block. EXEC is all-ones for every wave.
// ---------------------------------------------------------------------------
__global__ void __launch_bounds__(256) gemm16_kernel(
    const _Float16* __restrict__ A, long long sA, int lda,
    const _Float16* __restrict__ BT, long long sB, int ldb,
    int K, int rows, int cols,
    const float* __restrict__ bias, int relu,
    float* __restrict__ outF, long long sOF, int ldo,
    _Float16* __restrict__ outH, long long sOH,
    _Float16* __restrict__ outHT, long long sOHT, int ldto) {
  int wid = (blockIdx.x << 3) + (threadIdx.x >> 5);
  int colTiles = cols >> 4;
  int tilesPerB = (rows >> 4) * colTiles;
  int b = wid / tilesPerB;
  int t = wid % tilesPerB;
  int i0 = (t / colTiles) << 4;
  int j0 = (t % colTiles) << 4;
  const _Float16* Ab = A + (size_t)b * sA;
  const _Float16* Bb = BT + (size_t)b * sB;

  v8f acc = {};
  for (int k0 = 0; k0 < K; k0 += 32) {
    v16h av = load_a_tile(Ab, i0, k0, lda);
    v16h bv = load_b_tile(Bb, j0, k0, ldb);
    acc = __builtin_amdgcn_wmma_f32_16x16x32_f16(false, av, false, bv, (short)0, acc, false, false);
  }
  int lane = threadIdx.x & 31;
  int col  = j0 + (lane & 15);
  int rb   = i0 + ((lane & 16) ? 8 : 0);
  float bval = bias ? bias[col] : 0.0f;
#pragma unroll
  for (int v = 0; v < 8; ++v) {
    float val = acc[v] + bval;
    if (relu) val = fmaxf(val, 0.0f);
    int row = rb + v;
    if (outF)  outF [(size_t)b * sOF  + (size_t)row * ldo  + col] = val;
    if (outH)  outH [(size_t)b * sOH  + (size_t)row * ldo  + col] = (_Float16)val;
    if (outHT) outHT[(size_t)b * sOHT + (size_t)col * ldto + row] = (_Float16)val;
  }
}

// ---------------------------------------------------------------------------
// Prep: f16 transposed weight images + XW1 = emb @ W1 (shared across batch).
// ---------------------------------------------------------------------------
__global__ void __launch_bounds__(256) prep_kernel(
    const float* __restrict__ emb, const float* __restrict__ W1,
    const float* __restrict__ W2, const float* __restrict__ Wq,
    const float* __restrict__ Wk, const float* __restrict__ Wv,
    const float* __restrict__ Wo,
    _Float16* __restrict__ XW1T, _Float16* __restrict__ W2T,
    _Float16* __restrict__ WqT, _Float16* __restrict__ WkT,
    _Float16* __restrict__ WvT, _Float16* __restrict__ WoT) {
  int tid = threadIdx.x;
  // XW1T[f][c] = sum_e emb[c][e] * W1[e][f]   (f<128, c<128)
  for (int idx = tid; idx < 128 * 128; idx += 256) {
    int f = idx >> 7, c = idx & 127;
    float s = 0.0f;
    for (int e = 0; e < E_DIM; ++e) s += emb[c * E_DIM + e] * W1[e * 128 + f];
    XW1T[idx] = (_Float16)s;
  }
  // W2T[n][k] = W2[k][n]  (k<128, n<64)
  for (int idx = tid; idx < 64 * 128; idx += 256) {
    int n = idx >> 7, kx = idx & 127;
    W2T[idx] = (_Float16)W2[kx * 64 + n];
  }
  // 64x64 transposes for Wq/Wk/Wv/Wo
  for (int idx = tid; idx < 64 * 64; idx += 256) {
    int n = idx >> 6, kx = idx & 63;
    WqT[idx] = (_Float16)Wq[kx * 64 + n];
    WkT[idx] = (_Float16)Wk[kx * 64 + n];
    WvT[idx] = (_Float16)Wv[kx * 64 + n];
    WoT[idx] = (_Float16)Wo[kx * 64 + n];
  }
}

// ---------------------------------------------------------------------------
// Fused per-(batch,head) attention, online softmax (d=8, C=128 keys).
// Thread = one query row. Writes o as f16 rows for the final WMMA GEMM.
// ---------------------------------------------------------------------------
__global__ void __launch_bounds__(128) attn_kernel(const float* __restrict__ q,
                                                   const float* __restrict__ k,
                                                   const float* __restrict__ v,
                                                   _Float16* __restrict__ o) {
  __shared__ float ks[C_DIM][8], vs[C_DIM][8];
  int bh = blockIdx.x;
  int b = bh >> 3, h = bh & 7;
  size_t base = ((size_t)b * C_DIM) * E_DIM + h * 8;
  int c = threadIdx.x;
#pragma unroll
  for (int dd = 0; dd < 8; ++dd) {
    ks[c][dd] = k[base + (size_t)c * E_DIM + dd];
    vs[c][dd] = v[base + (size_t)c * E_DIM + dd];
  }
  __syncthreads();
  float qr[8];
#pragma unroll
  for (int dd = 0; dd < 8; ++dd) qr[dd] = q[base + (size_t)c * E_DIM + dd];

  const float scale = 0.35355339059327373f;  // 8^-0.5
  float m = -1e30f, l = 0.0f, acc[8] = {0, 0, 0, 0, 0, 0, 0, 0};
  for (int j = 0; j < C_DIM; ++j) {
    float s = 0.0f;
#pragma unroll
    for (int dd = 0; dd < 8; ++dd) s += qr[dd] * ks[j][dd];
    s *= scale;
    float mn = fmaxf(m, s);
    float corr = __expf(m - mn);
    float w = __expf(s - mn);
    l = l * corr + w;
#pragma unroll
    for (int dd = 0; dd < 8; ++dd) acc[dd] = acc[dd] * corr + w * vs[j][dd];
    m = mn;
  }
  float invl = 1.0f / l;
#pragma unroll
  for (int dd = 0; dd < 8; ++dd)
    o[base + (size_t)c * E_DIM + dd] = (_Float16)(acc[dd] * invl);
}

// ---------------------------------------------------------------------------
extern "C" void kernel_launch(void* const* d_in, const int* in_sizes, int n_in,
                              void* d_out, int out_size, void* d_ws, size_t ws_size,
                              hipStream_t stream) {
  const float* x   = (const float*)d_in[0];
  const float* emb = (const float*)d_in[1];
  const float* W1  = (const float*)d_in[2];
  const float* b1  = (const float*)d_in[3];
  const float* W2  = (const float*)d_in[4];
  const float* b2  = (const float*)d_in[5];
  const float* Wq  = (const float*)d_in[6];
  const float* bq  = (const float*)d_in[7];
  const float* Wk  = (const float*)d_in[8];
  const float* bk  = (const float*)d_in[9];
  const float* Wv  = (const float*)d_in[10];
  const float* bv  = (const float*)d_in[11];
  const float* Wo  = (const float*)d_in[12];
  const float* bo  = (const float*)d_in[13];

  float* out       = (float*)d_out;
  float* out_conn  = out;                                        // B*C*C
  float* out_graph = out + (size_t)B_DIM * C_DIM * C_DIM;        // B*C*E
  float* out_h2    = out_graph + (size_t)B_DIM * C_DIM * E_DIM;  // B*C*E

  // workspace carve-up (deterministic, 256B aligned)
  char* w = (char*)d_ws;
  auto carve = [&](size_t bytes) -> char* {
    char* p = w;
    w += (bytes + 255) & ~(size_t)255;
    return p;
  };
  const size_t BCT = (size_t)B_DIM * C_DIM * T_DIM;
  const size_t BCC = (size_t)B_DIM * C_DIM * C_DIM;
  const size_t BCE = (size_t)B_DIM * C_DIM * E_DIM;
  _Float16* zc      = (_Float16*)carve(BCT * 2);
  _Float16* zs      = (_Float16*)carve(BCT * 2);
  _Float16* conn_h  = (_Float16*)carve(BCC * 2);
  _Float16* XW1T    = (_Float16*)carve(128 * 128 * 2);
  _Float16* h1_h    = (_Float16*)carve((size_t)B_DIM * C_DIM * 128 * 2);
  _Float16* W2T     = (_Float16*)carve(64 * 128 * 2);
  _Float16* YT      = (_Float16*)carve((size_t)B_DIM * 64 * 128 * 2);
  _Float16* h2_h    = (_Float16*)carve(BCE * 2);
  _Float16* WqT     = (_Float16*)carve(64 * 64 * 2);
  _Float16* WkT     = (_Float16*)carve(64 * 64 * 2);
  _Float16* WvT     = (_Float16*)carve(64 * 64 * 2);
  _Float16* WoT     = (_Float16*)carve(64 * 64 * 2);
  float*    qf      = (float*)carve(BCE * 4);
  float*    kf      = (float*)carve(BCE * 4);
  float*    vf      = (float*)carve(BCE * 4);
  _Float16* o_h     = (_Float16*)carve(BCE * 2);
  (void)ws_size; (void)in_sizes; (void)n_in; (void)out_size;

  // 1) Hilbert phasors
  hipLaunchKernelGGL(hilbert_kernel, dim3(B_DIM * C_DIM), dim3(256), 0, stream, x, zc, zs);

  // 2) weight prep
  hipLaunchKernelGGL(prep_kernel, dim3(1), dim3(256), 0, stream,
                     emb, W1, W2, Wq, Wk, Wv, Wo, XW1T, W2T, WqT, WkT, WvT, WoT);

  // 3) connectivity (WMMA)  : 8192 tiles / 8 waves per block
  hipLaunchKernelGGL(conn_kernel, dim3(1024), dim3(256), 0, stream, zc, zs, out_conn, conn_h);

  // 4) h1 = relu(conn @ XW1 + b1)      (128x128, K=128)  8192 tiles
  hipLaunchKernelGGL(gemm16_kernel, dim3(1024), dim3(256), 0, stream,
                     conn_h, (long long)(C_DIM * C_DIM), C_DIM,
                     XW1T, 0LL, 128,
                     128, 128, 128,
                     b1, 1,
                     (float*)nullptr, 0LL, 128,
                     h1_h, (long long)(128 * 128),
                     (_Float16*)nullptr, 0LL, 0);

  // 5) Y = h1 @ W2  -> store Y^T       (128x64, K=128)   4096 tiles
  hipLaunchKernelGGL(gemm16_kernel, dim3(512), dim3(256), 0, stream,
                     h1_h, (long long)(128 * 128), 128,
                     W2T, 0LL, 128,
                     128, 128, 64,
                     (const float*)nullptr, 0,
                     (float*)nullptr, 0LL, 64,
                     (_Float16*)nullptr, 0LL,
                     YT, (long long)(64 * 128), 128);

  // 6) h2 = conn @ Y + b2              (128x64, K=128)   4096 tiles
  hipLaunchKernelGGL(gemm16_kernel, dim3(512), dim3(256), 0, stream,
                     conn_h, (long long)(C_DIM * C_DIM), C_DIM,
                     YT, (long long)(64 * 128), 128,
                     128, 128, 64,
                     b2, 0,
                     out_h2, (long long)(C_DIM * E_DIM), 64,
                     h2_h, (long long)(C_DIM * E_DIM),
                     (_Float16*)nullptr, 0LL, 0);

  // 7) q/k/v = h2 @ W{q,k,v} + b       (128x64, K=64)
  const _Float16* WT3[3] = { WqT, WkT, WvT };
  const float*    bb3[3] = { bq, bk, bv };
  float*          oo3[3] = { qf, kf, vf };
  for (int p = 0; p < 3; ++p) {
    hipLaunchKernelGGL(gemm16_kernel, dim3(512), dim3(256), 0, stream,
                       h2_h, (long long)(C_DIM * E_DIM), 64,
                       WT3[p], 0LL, 64,
                       64, 128, 64,
                       bb3[p], 0,
                       oo3[p], (long long)(C_DIM * E_DIM), 64,
                       (_Float16*)nullptr, 0LL,
                       (_Float16*)nullptr, 0LL, 0);
  }

  // 8) fused multi-head attention -> o (f16)
  hipLaunchKernelGGL(attn_kernel, dim3(B_DIM * 8), dim3(128), 0, stream, qf, kf, vf, o_h);

  // 9) graph_features = o @ Wo + bo    (128x64, K=64)
  hipLaunchKernelGGL(gemm16_kernel, dim3(512), dim3(256), 0, stream,
                     o_h, (long long)(C_DIM * E_DIM), 64,
                     WoT, 0LL, 64,
                     64, 128, 64,
                     bo, 0,
                     out_graph, (long long)(C_DIM * E_DIM), 64,
                     (_Float16*)nullptr, 0LL,
                     (_Float16*)nullptr, 0LL, 0);
}
